// EuclideanAttentionBlock_53154515255878
// MI455X (gfx1250) — matrix operand order repaired and checked by reading, not compiled, not run on hardware
//
#include <hip/hip_runtime.h>
#include <hip/hip_bf16.h>

typedef __attribute__((ext_vector_type(16))) _Float16 v16h;
typedef __attribute__((ext_vector_type(8)))  _Float16 v8h;
typedef __attribute__((ext_vector_type(8)))  float    v8f;

#define F_DIM   128
#define HID     128
#define EVD     15
#define NRBF    32
#define MAXL    3
#define DIN     35      // 32 + 3
#define KPAD1   64      // padded K for GEMM1
#define XS      72      // X row stride (halves), padded
#define WS1     72      // W1^T row stride (halves)
#define WS2     136     // W2^T / H row stride (halves)
#define MTILE   128
#define ITERS   8
#define EPB     (MTILE * ITERS)

// ---- WMMA fragment loaders (ISA 7.12.2 layouts, wave32) ----
// A (16x32 f16): lanes 0-15 -> M=lane, K in {k0..k0+7, k0+16..k0+23}
//                lanes 16-31 -> M=lane-16, K in {k0+8..k0+15, k0+24..k0+31}
__device__ __forceinline__ v16h load_a_frag(const _Float16* base, int stride,
                                            int m0, int k0, int lane) {
    int row = m0 + (lane & 15);
    int khi = (lane >> 4) & 1;
    const _Float16* p = base + row * stride + k0 + khi * 8;
    v8h lo = *(const v8h*)(p);
    v8h hi = *(const v8h*)(p + 16);
    v16h a;
#pragma unroll
    for (int i = 0; i < 8; ++i) { a[i] = lo[i]; a[i + 8] = hi[i]; }
    return a;
}

// B (32x16 f16), weights stored transposed in LDS as WT[n][k]:
// lanes 0-15 -> N=lane, K in {k0..k0+7, k0+16..k0+23} ; lanes 16-31 -> N=lane-16, K in {k0+8..k0+15, k0+24..k0+31}
__device__ __forceinline__ v16h load_b_frag(const _Float16* baseT, int stride,
                                            int n0, int k0, int lane) {
    int row = n0 + (lane & 15);
    int kb  = k0 + ((lane >> 4) & 1) * 8;
    const _Float16* p = baseT + row * stride + kb;
    v8h lo = *(const v8h*)(p);
    v8h hi = *(const v8h*)(p + 16);
    v16h b;
#pragma unroll
    for (int i = 0; i < 8; ++i) { b[i] = lo[i]; b[i + 8] = hi[i]; }
    return b;
}

// silu via v_rcp_f32 (avoids the IEEE v_div_scale/div_fmas expansion)
__device__ __forceinline__ float silu(float x) {
    return x * __builtin_amdgcn_rcpf(1.0f + __expf(-x));
}

__device__ __forceinline__ unsigned pack_f16x2(float a, float b) {
    union { _Float16 h[2]; unsigned u; } v;
    v.h[0] = (_Float16)a;
    v.h[1] = (_Float16)b;
    return v.u;
}

// Stage weight matrix W[K,N] (fp32, row-major) into LDS as f16 WT[n][k],
// zero-padding to Kpad x Nrows. Two K-adjacent values packed per b32 store.
template <int K, int N, int Kpad, int Nrows, int STRIDE>
__device__ __forceinline__ void stage_weightT(const float* __restrict__ W,
                                              _Float16* dst, int tid) {
    constexpr int total = (Kpad / 2) * Nrows;
    for (int idx = tid; idx < total; idx += 256) {
        int n  = idx % Nrows;          // consecutive tids -> consecutive n (coalesced gmem)
        int k2 = idx / Nrows;
        int k  = 2 * k2;
        float v0 = (k     < K && n < N) ? W[(k)     * N + n] : 0.0f;
        float v1 = (k + 1 < K && n < N) ? W[(k + 1) * N + n] : 0.0f;
        *(unsigned*)(dst + n * STRIDE + k) = pack_f16x2(v0, v1);
    }
}

__global__ __launch_bounds__(256)
void edge_filter_mlp_kernel(const float* __restrict__ ev,
                            const int*   __restrict__ senders,
                            const int*   __restrict__ receivers,
                            const float* __restrict__ lengths,
                            const float* __restrict__ W1i, const float* __restrict__ b1i,
                            const float* __restrict__ W2i, const float* __restrict__ b2i,
                            const float* __restrict__ W1e, const float* __restrict__ b1e,
                            const float* __restrict__ W2e, const float* __restrict__ b2e,
                            float* __restrict__ out_inv,
                            float* __restrict__ out_ev,
                            int E) {
    __shared__ __align__(16) _Float16 sX  [MTILE * XS];     // 18 KB
    __shared__ __align__(16) _Float16 sW1i[HID   * WS1];    // 18 KB
    __shared__ __align__(16) _Float16 sW1e[HID   * WS1];    // 18 KB
    __shared__ __align__(16) _Float16 sW2i[F_DIM * WS2];    // 34 KB
    __shared__ __align__(16) _Float16 sW2e[16    * WS2];    // 4.25 KB
    __shared__ __align__(16) _Float16 sH  [MTILE * WS2];    // 34 KB
    __shared__ float sb1i[HID], sb1e[HID], sb2i[F_DIM], sb2e[16];

    const int tid  = threadIdx.x;
    const int lane = tid & 31;
    const int wave = tid >> 5;
    const int m0   = wave * 16;          // 8 waves cover M=128

    // ---- stage all weights/biases once per block ----
    stage_weightT<DIN, HID,  KPAD1, HID,  WS1>(W1i, sW1i, tid);
    stage_weightT<DIN, HID,  KPAD1, HID,  WS1>(W1e, sW1e, tid);
    stage_weightT<HID, F_DIM, HID,  F_DIM, WS2>(W2i, sW2i, tid);
    stage_weightT<HID, MAXL,  HID,  16,    WS2>(W2e, sW2e, tid);
    if (tid < HID) { sb1i[tid] = b1i[tid]; sb1e[tid] = b1e[tid]; sb2i[tid] = b2i[tid]; }
    if (tid < 16)  { sb2e[tid] = (tid < MAXL) ? b2e[tid] : 0.0f; }

    const long long blockEdge0 = (long long)blockIdx.x * EPB;
    const int nlocal = lane & 15;
    const int rowoff = ((lane >> 4) & 1) * 8;   // C-matrix: lanes>=16 hold M=r+8

    for (int it = 0; it < ITERS; ++it) {
        const long long edge0 = blockEdge0 + (long long)it * MTILE;

        // ---- build X tile: [MTILE][64] f16 = [lengths(32) | inv(3) | 0-pad] ----
        {
            const int m    = tid & 127;
            const int part = tid >> 7;
            const long long e = edge0 + m;
            const bool valid  = (e < (long long)E);
            _Float16* row = sX + m * XS;
            if (part == 0) {
                if (valid) {
                    const float4* Lp = (const float4*)(lengths + (size_t)e * NRBF);
#pragma unroll
                    for (int i = 0; i < 8; ++i) {
                        float4 v = Lp[i];
                        *(unsigned*)(row + 4 * i)     = pack_f16x2(v.x, v.y);
                        *(unsigned*)(row + 4 * i + 2) = pack_f16x2(v.z, v.w);
                    }
                } else {
#pragma unroll
                    for (int i = 0; i < 16; ++i) *(unsigned*)(row + 2 * i) = 0u;
                }
            } else {
                float i1 = 0.0f, i2 = 0.0f, i3 = 0.0f;
                if (valid) {
                    const int s = senders[e];
                    const int r = receivers[e];
                    const float* ps = ev + (size_t)s * EVD;
                    const float* pr = ev + (size_t)r * EVD;
#pragma unroll
                    for (int j = 0; j < 3; ++j)  { float d = ps[j] - pr[j]; i1 += d * d; }
#pragma unroll
                    for (int j = 3; j < 8; ++j)  { float d = ps[j] - pr[j]; i2 += d * d; }
#pragma unroll
                    for (int j = 8; j < 15; ++j) { float d = ps[j] - pr[j]; i3 += d * d; }
                }
                *(unsigned*)(row + 32) = pack_f16x2(i1, i2);
                *(unsigned*)(row + 34) = pack_f16x2(i3, 0.0f);
#pragma unroll
                for (int j = 18; j < 32; ++j) *(unsigned*)(row + 2 * j) = 0u;
            }
        }
        __syncthreads();   // X ready (also guards sH reads of previous iter)

        // ================= inv head =================
        // GEMM1: H = silu(X @ W1i + b1i)   [128x64]x[64x128]
        {
            v16h a0 = load_a_frag(sX, XS, m0, 0,  lane);
            v16h a1 = load_a_frag(sX, XS, m0, 32, lane);
#pragma unroll
            for (int nt = 0; nt < 8; ++nt) {
                const int n0 = nt * 16;
                v16h b0 = load_b_frag(sW1i, WS1, n0, 0,  lane);
                v16h b1 = load_b_frag(sW1i, WS1, n0, 32, lane);
                v8f acc = {};
                acc = __builtin_amdgcn_wmma_f32_16x16x32_f16(false, a0, false, b0, (short)0, acc, false, false);
                acc = __builtin_amdgcn_wmma_f32_16x16x32_f16(false, a1, false, b1, (short)0, acc, false, false);
                const float bias = sb1i[n0 + nlocal];
#pragma unroll
                for (int r = 0; r < 8; ++r) {
                    sH[(m0 + r + rowoff) * WS2 + n0 + nlocal] = (_Float16)silu(acc[r] + bias);
                }
            }
        }
        __syncthreads();   // H ready

        // GEMM2: out_inv = H @ W2i + b2i   [128x128]x[128x128]
        {
            v16h a[4];
#pragma unroll
            for (int kk = 0; kk < 4; ++kk) a[kk] = load_a_frag(sH, WS2, m0, kk * 32, lane);
#pragma unroll
            for (int nt = 0; nt < 8; ++nt) {
                const int n0 = nt * 16;
                v8f acc = {};
#pragma unroll
                for (int kk = 0; kk < 4; ++kk) {
                    v16h b = load_b_frag(sW2i, WS2, n0, kk * 32, lane);
                    acc = __builtin_amdgcn_wmma_f32_16x16x32_f16(false, a[kk], false, b, (short)0, acc, false, false);
                }
                const float bias = sb2i[n0 + nlocal];
#pragma unroll
                for (int r = 0; r < 8; ++r) {
                    const long long e = edge0 + m0 + r + rowoff;
                    if (e < (long long)E) {
                        out_inv[(size_t)e * F_DIM + n0 + nlocal] = acc[r] + bias;
                    }
                }
            }
        }
        __syncthreads();   // sH reads done before overwrite

        // ================= ev head =================
        // GEMM1: H = silu(X @ W1e + b1e)
        {
            v16h a0 = load_a_frag(sX, XS, m0, 0,  lane);
            v16h a1 = load_a_frag(sX, XS, m0, 32, lane);
#pragma unroll
            for (int nt = 0; nt < 8; ++nt) {
                const int n0 = nt * 16;
                v16h b0 = load_b_frag(sW1e, WS1, n0, 0,  lane);
                v16h b1 = load_b_frag(sW1e, WS1, n0, 32, lane);
                v8f acc = {};
                acc = __builtin_amdgcn_wmma_f32_16x16x32_f16(false, a0, false, b0, (short)0, acc, false, false);
                acc = __builtin_amdgcn_wmma_f32_16x16x32_f16(false, a1, false, b1, (short)0, acc, false, false);
                const float bias = sb1e[n0 + nlocal];
#pragma unroll
                for (int r = 0; r < 8; ++r) {
                    sH[(m0 + r + rowoff) * WS2 + n0 + nlocal] = (_Float16)silu(acc[r] + bias);
                }
            }
        }
        __syncthreads();   // H ready

        // GEMM2: out_ev = H @ W2e + b2e   [128x128]x[128x3] (N padded to 16)
        {
            v16h a[4];
#pragma unroll
            for (int kk = 0; kk < 4; ++kk) a[kk] = load_a_frag(sH, WS2, m0, kk * 32, lane);
            v8f acc = {};
#pragma unroll
            for (int kk = 0; kk < 4; ++kk) {
                v16h b = load_b_frag(sW2e, WS2, 0, kk * 32, lane);
                acc = __builtin_amdgcn_wmma_f32_16x16x32_f16(false, a[kk], false, b, (short)0, acc, false, false);
            }
            if (nlocal < MAXL) {
                const float bias = sb2e[nlocal];
#pragma unroll
                for (int r = 0; r < 8; ++r) {
                    const long long e = edge0 + m0 + r + rowoff;
                    if (e < (long long)E) {
                        out_ev[(size_t)e * MAXL + nlocal] = acc[r] + bias;
                    }
                }
            }
        }
        __syncthreads();   // before next-iter X rebuild / H overwrite
    }
}

extern "C" void kernel_launch(void* const* d_in, const int* in_sizes, int n_in,
                              void* d_out, int out_size, void* d_ws, size_t ws_size,
                              hipStream_t stream) {
    (void)n_in; (void)out_size; (void)d_ws; (void)ws_size;
    // setup_inputs order:
    // 0 inv_features, 1 ev_features, 2 senders, 3 receivers, 4 sh_vectors,
    // 5 lengths, 6 cutoffs, 7 W1_inv, 8 b1_inv, 9 W2_inv, 10 b2_inv,
    // 11 W1_ev, 12 b1_ev, 13 W2_ev, 14 b2_ev
    const float* ev        = (const float*)d_in[1];
    const int*   senders   = (const int*)d_in[2];
    const int*   receivers = (const int*)d_in[3];
    const float* lengths   = (const float*)d_in[5];
    const float* W1i = (const float*)d_in[7];
    const float* b1i = (const float*)d_in[8];
    const float* W2i = (const float*)d_in[9];
    const float* b2i = (const float*)d_in[10];
    const float* W1e = (const float*)d_in[11];
    const float* b1e = (const float*)d_in[12];
    const float* W2e = (const float*)d_in[13];
    const float* b2e = (const float*)d_in[14];

    const int E = in_sizes[2];
    float* out_inv = (float*)d_out;
    float* out_ev  = out_inv + (size_t)E * F_DIM;

    const int nblocks = (E + EPB - 1) / EPB;
    edge_filter_mlp_kernel<<<dim3(nblocks), dim3(256), 0, stream>>>(
        ev, senders, receivers, lengths,
        W1i, b1i, W2i, b2i, W1e, b1e, W2e, b2e,
        out_inv, out_ev, E);
}
